// SemanticComm_64433099375345
// MI455X (gfx1250) — matrix operand order repaired and checked by reading, not compiled, not run on hardware
//
#include <hip/hip_runtime.h>
#include <hip/hip_bf16.h>

// ---------------------------------------------------------------------------
// Types for CDNA5 WMMA (wave32, 16x16x32 bf16 -> f32)
// ---------------------------------------------------------------------------
typedef __attribute__((ext_vector_type(16))) __bf16        v16bf;
typedef __attribute__((ext_vector_type(8)))  float         v8f;
typedef __attribute__((ext_vector_type(4)))  unsigned int  v4u;

union AFrag { v16bf v; v4u q[2]; };

__device__ __forceinline__ unsigned short f2bf(float f) {
    unsigned int x = __builtin_bit_cast(unsigned int, f);
    unsigned int r = (x + 0x7FFFu + ((x >> 16) & 1u)) >> 16;   // RNE
    return (unsigned short)r;
}

#define B_TOT   1024
#define DIN     3072
#define DZ      512
#define DOUT    3072
#define N_SYM   8
#define NSC     32
#define MFFT    64
#define LENCP   16
#define LEN_T   640
#define LENRRC  129
#define L1N     5248
#define TSTART  65
#define PWIN    5120
#define OUT_PAPR_OFF 3145728   // 1024*3072
#define OUT_LOSS_OFF 3146752   // + 1024

// ---------------------------------------------------------------------------
// Pack kernel A: f32 -> bf16, same linear layout (2 elements / thread).
// ---------------------------------------------------------------------------
__global__ __launch_bounds__(256) void pack_linear_bf16(
    const float* __restrict__ src, unsigned int* __restrict__ dst, int nWords)
{
    int i = blockIdx.x * 256 + threadIdx.x;
    if (i >= nWords) return;
    const float2 v = ((const float2*)src)[i];
    dst[i] = (unsigned int)f2bf(v.x) | ((unsigned int)f2bf(v.y) << 16);
}

// ---------------------------------------------------------------------------
// Pack kernel B: W(KxN f32, row-major) -> bf16 tiles [kblk][n][32 K-values]
// so a WMMA B-fragment is 2 contiguous 16B vector loads per lane.
// ---------------------------------------------------------------------------
__global__ __launch_bounds__(256) void pack_w_tiles(
    const float* __restrict__ W, unsigned int* __restrict__ dst, int K, int N)
{
    size_t i = (size_t)blockIdx.x * 256 + threadIdx.x;       // u32 word index
    size_t total = (size_t)K * N / 2;
    if (i >= total) return;
    size_t sb   = 2 * i;                                     // short index
    int    q0   = (int)(sb & 15);                            // even
    int    h    = (int)((sb >> 4) & 1);
    size_t n    = (sb >> 5) % (size_t)N;
    size_t kblk = (sb >> 5) / (size_t)N;
    size_t k    = kblk * 32 + (size_t)h * 16 + q0;
    unsigned int lo = f2bf(W[k * N + n]);
    unsigned int hi = f2bf(W[(k + 1) * N + n]);
    dst[i] = lo | (hi << 16);
}

// ---------------------------------------------------------------------------
// Kernel 1: encoder GEMM (bf16 WMMA) + power-norm.
// grid 64 blocks x 256 thr. Each block: 16 rows x 512 cols, K=3072.
// A,B pre-packed bf16 -> fragments load directly via b128, no LDS in k-loop.
// ---------------------------------------------------------------------------
__global__ __launch_bounds__(256) void enc_gemm(
    const unsigned short* __restrict__ packX,
    const unsigned short* __restrict__ packW,
    const float* __restrict__ bias, float* __restrict__ zout)
{
    __shared__ float cTile[16][DZ];
    __shared__ float rowMean[16];
    __shared__ float rowInvStd[16];
    __shared__ float red[256];

    const int m0    = blockIdx.x * 16;
    const int tid   = threadIdx.x;
    const int wave  = tid >> 5;
    const int lane  = tid & 31;
    const int lrow  = lane & 15;
    const bool hi   = lane >= 16;
    const int nbase = wave * 64;
    const int koffA = hi ? 16 : 0;
    const int hB    = hi ? 16 : 0;

    v8f acc0 = {}, acc1 = {}, acc2 = {}, acc3 = {};
    v8f* acc[4] = {&acc0, &acc1, &acc2, &acc3};

    const unsigned short* aBase = packX + (size_t)(m0 + lrow) * DIN + koffA;

    for (int kk = 0; kk < DIN / 32; ++kk) {
        const int k0 = kk * 32;
        AFrag a;
        {
            const v4u* ap = (const v4u*)(aBase + k0);  // K{koffA..+15}
            a.q[0] = ap[0];
            a.q[1] = ap[1];
            if (kk + 1 < DIN / 32) __builtin_prefetch(ap + 4, 0, 1);
        }
#pragma unroll
        for (int t = 0; t < 4; ++t) {
            const int col = nbase + t * 16 + lrow;
            const v4u* bp =
                (const v4u*)(packW + (((size_t)kk * DZ + col) << 5) + hB);
            AFrag bf;
            bf.q[0] = bp[0];
            bf.q[1] = bp[1];
            *acc[t] = __builtin_amdgcn_wmma_f32_16x16x32_bf16(
                false, a.v, false, bf.v, (short)0, *acc[t], false, false);
        }
    }

    // C/D layout: lane<16 -> M=v, lane>=16 -> M=v+8 ; N = nbase+t*16+(lane&15)
#pragma unroll
    for (int t = 0; t < 4; ++t) {
        const int col = nbase + t * 16 + lrow;
        const float bv = bias[col];
#pragma unroll
        for (int v = 0; v < 8; ++v) {
            const int row = hi ? (v + 8) : v;
            cTile[row][col] = (*acc[t])[v] + bv;
        }
    }
    __syncthreads();

    // power-norm: 16 threads per row, 32 cols each
    const int r   = tid >> 4;
    const int seg = tid & 15;
    float s = 0.f;
    for (int c = seg * 32; c < seg * 32 + 32; ++c) s += cTile[r][c];
    red[tid] = s;
    __syncthreads();
    if (seg == 0) {
        float m = 0.f;
        for (int i = 0; i < 16; ++i) m += red[(r << 4) + i];
        rowMean[r] = m / 512.f;
    }
    __syncthreads();
    const float mean = rowMean[r];
    float vv = 0.f;
    for (int c = seg * 32; c < seg * 32 + 32; ++c) {
        float d = cTile[r][c] - mean; vv += d * d;
    }
    red[tid] = vv;
    __syncthreads();
    if (seg == 0) {
        float vs = 0.f;
        for (int i = 0; i < 16; ++i) vs += red[(r << 4) + i];
        rowInvStd[r] = 1.f / (sqrtf(vs / 511.f) + 1e-8f);   // ddof=1
    }
    __syncthreads();
    const float is = rowInvStd[r];
    for (int c = seg * 32; c < seg * 32 + 32; ++c)
        zout[(size_t)(m0 + r) * DZ + c] = (cTile[r][c] - mean) * is;
}

// ---------------------------------------------------------------------------
// Kernel 2: per-batch-row OFDM / RRC / carrier / clip / PAPR / AWGN / RX chain.
// grid 1024 blocks x 256 thr; everything in LDS (~38 KB). Output: packed bf16.
// ---------------------------------------------------------------------------
__global__ __launch_bounds__(256) void dsp_pipeline(
    const float* __restrict__ zbuf, const float* __restrict__ rrc,
    const float* __restrict__ awgn, const int* __restrict__ startloc,
    unsigned int* __restrict__ ybuf, float* __restrict__ paprOut,
    float* __restrict__ reluOut)
{
    __shared__ float  rrcL[LENRRC];
    __shared__ float  tc[64], ts[64];          // cos/sin(2*pi*r/64)
    __shared__ float  cc16[16], ss16[16];      // sqrt(2)*cos/sin(2*pi*5*r/16)
    __shared__ float2 cb[256];
    __shared__ float2 Cb[256];
    __shared__ float2 dT[LEN_T];
    __shared__ float  dx[L1N];
    __shared__ float2 y6[LEN_T];
    __shared__ float  red[256], redM[256];
    __shared__ float  sc[1];

    const int b   = blockIdx.x;
    const int tid = threadIdx.x;
    const int sl  = startloc[0];
    const float TWO_PI = 6.283185307179586f;

    for (int i = tid; i < LENRRC; i += 256) rrcL[i] = rrc[i];
    for (int i = tid; i < 64; i += 256) {
        float a = TWO_PI * (float)i / 64.f;
        tc[i] = cosf(a); ts[i] = sinf(a);
    }
    for (int i = tid; i < 16; i += 256) {
        float a = TWO_PI * 5.f * (float)i / 16.f;
        cc16[i] = 1.41421356237f * cosf(a);
        ss16[i] = 1.41421356237f * sinf(a);
    }
    {
        const float* zr = zbuf + (size_t)b * DZ;
        cb[tid] = make_float2(zr[2 * tid], zr[2 * tid + 1]);
    }
    __syncthreads();

    // ---- TX DFT-32 (ortho) ----
    {
        const int s = tid >> 5, j = tid & 31;
        float ar = 0.f, ai = 0.f;
        for (int n = 0; n < 32; ++n) {
            int r = (j * n) & 31;
            float wr = tc[2 * r], wi = -ts[2 * r];
            float2 v = cb[s * 32 + n];
            ar += v.x * wr - v.y * wi;
            ai += v.x * wi + v.y * wr;
        }
        Cb[tid] = make_float2(ar * 0.1767766953f, ai * 0.1767766953f);
    }
    __syncthreads();

    // ---- embed at loc=[sl,sl+32) with fftshift, IDFT-64 (ortho), add CP ----
    for (int i = tid; i < N_SYM * MFFT; i += 256) {
        const int s = i >> 6, m = i & 63;
        float ar = 0.f, ai = 0.f;
        for (int k = sl; k < sl + NSC; ++k) {
            int cidx = ((k - sl) + 16) & 31;          // fftshift
            float2 c = Cb[s * 32 + cidx];
            int r = (k * m) & 63;
            float wr = tc[r], wi = ts[r];             // e^{+i}
            ar += c.x * wr - c.y * wi;
            ai += c.x * wi + c.y * wr;
        }
        float2 val = make_float2(ar * 0.125f, ai * 0.125f);
        dT[s * 80 + LENCP + m] = val;
        if (m >= MFFT - LENCP) dT[s * 80 + (m - (MFFT - LENCP))] = val;
    }
    __syncthreads();

    // ---- x8 upsample + RRC (polyphase full conv) + carrier up-mix ----
    for (int n = tid; n < L1N; n += 256) {
        int jlo = (n >= 128) ? ((n - 128 + 7) >> 3) : 0;
        int jhi = n >> 3; if (jhi > LEN_T - 1) jhi = LEN_T - 1;
        float xr = 0.f, xi = 0.f;
        for (int j = jlo; j <= jhi; ++j) {
            float tap = rrcL[n - 8 * j];
            xr += dT[j].x * tap;
            xi += dT[j].y * tap;
        }
        int r = n & 15;
        dx[n] = xr * cc16[r] - xi * ss16[r];
    }
    __syncthreads();

    // ---- soft clip: thr = 1.2*rms ----
    {
        float p = 0.f;
        for (int n = tid; n < L1N; n += 256) p += dx[n] * dx[n];
        red[tid] = p; __syncthreads();
        for (int off = 128; off > 0; off >>= 1) {
            if (tid < off) red[tid] += red[tid + off];
            __syncthreads();
        }
        if (tid == 0) sc[0] = 1.2f * sqrtf(red[0] / (float)L1N);
        __syncthreads();
        const float thr = sc[0];
        for (int n = tid; n < L1N; n += 256) {
            float v = dx[n], av = fabsf(v);
            float oa = fmaxf(av - thr, 0.f);
            dx[n] = (1.f - oa / (av + 1e-8f)) * v;
        }
        __syncthreads();
    }

    // ---- PAPR over [TSTART, TSTART+PWIN) ----
    {
        float ps = 0.f, pm = 0.f;
        for (int n = TSTART + tid; n < TSTART + PWIN; n += 256) {
            float p = dx[n] * dx[n];
            ps += p; pm = fmaxf(pm, p);
        }
        red[tid] = ps; redM[tid] = pm; __syncthreads();
        for (int off = 128; off > 0; off >>= 1) {
            if (tid < off) {
                red[tid]  += red[tid + off];
                redM[tid]  = fmaxf(redM[tid], redM[tid + off]);
            }
            __syncthreads();
        }
        if (tid == 0) {
            float mean = red[0] / (float)PWIN;
            float papr = 10.f * log10f(redM[0] / mean);
            paprOut[b] = papr;
            reluOut[b] = fmaxf(papr - 8.f, 0.f);
        }
        __syncthreads();
    }

    // ---- AWGN ----
    for (int n = tid; n < L1N; n += 256) dx[n] += awgn[(size_t)b * L1N + n];
    __syncthreads();

    // ---- down-mix + RRC matched filter, only the 640 subsampled taps ----
    for (int u = tid; u < LEN_T; u += 256) {
        float ar = 0.f, ai = 0.f;
        for (int j = 0; j < LENRRC; ++j) {
            int n = 128 + 8 * u - j;
            float d = dx[n] * rrcL[j];
            int r = n & 15;
            ar += d * cc16[r];
            ai -= d * ss16[r];
        }
        y6[u] = make_float2(ar, ai);
    }
    __syncthreads();

    // ---- strip CP, DFT-64 (ortho), keep bins loc, fftshift ----
    {
        const int s = tid >> 5, j = tid & 31;
        const int k = sl + ((j + 16) & 31);
        float ar = 0.f, ai = 0.f;
        for (int m = 0; m < MFFT; ++m) {
            float2 v = y6[s * 80 + LENCP + m];
            int r = (k * m) & 63;
            float wr = tc[r], wi = -ts[r];
            ar += v.x * wr - v.y * wi;
            ai += v.x * wi + v.y * wr;
        }
        Cb[tid] = make_float2(ar * 0.125f, ai * 0.125f);
    }
    __syncthreads();

    // ---- IDFT-32 (ortho), emit packed bf16 (re,im) for the decoder GEMM ----
    {
        const int s = tid >> 5, n = tid & 31;
        float ar = 0.f, ai = 0.f;
        for (int j = 0; j < 32; ++j) {
            float2 v = Cb[s * 32 + j];
            int r = (j * n) & 31;
            float wr = tc[2 * r], wi = ts[2 * r];
            ar += v.x * wr - v.y * wi;
            ai += v.x * wi + v.y * wr;
        }
        unsigned int w = (unsigned int)f2bf(ar * 0.1767766953f) |
                         ((unsigned int)f2bf(ai * 0.1767766953f) << 16);
        ybuf[(size_t)b * 256 + tid] = w;
    }
}

// ---------------------------------------------------------------------------
// Kernel 3: decoder GEMM (bf16 WMMA). grid 64x6=384 blocks, 16 rows x 512 cols,
// K = 512. A = packed bf16 from dsp, B = pre-packed tiles. No LDS at all.
// ---------------------------------------------------------------------------
__global__ __launch_bounds__(256) void dec_gemm(
    const unsigned short* __restrict__ packY,
    const unsigned short* __restrict__ packW,
    const float* __restrict__ bias, float* __restrict__ out)
{
    const int mblk = blockIdx.x / 6;
    const int nblk = blockIdx.x % 6;
    const int m0   = mblk * 16;
    const int nb0  = nblk * 512;
    const int tid  = threadIdx.x;
    const int wave = tid >> 5;
    const int lane = tid & 31;
    const int lrow = lane & 15;
    const bool hi  = lane >= 16;
    const int nbase = wave * 64;
    const int koffA = hi ? 16 : 0;
    const int hB    = hi ? 16 : 0;

    v8f acc0 = {}, acc1 = {}, acc2 = {}, acc3 = {};
    v8f* acc[4] = {&acc0, &acc1, &acc2, &acc3};

    const unsigned short* aBase = packY + (size_t)(m0 + lrow) * DZ + koffA;

    for (int kk = 0; kk < DZ / 32; ++kk) {
        const int k0 = kk * 32;
        AFrag a;
        {
            const v4u* ap = (const v4u*)(aBase + k0);
            a.q[0] = ap[0];
            a.q[1] = ap[1];
        }
#pragma unroll
        for (int t = 0; t < 4; ++t) {
            const int gcol = nb0 + nbase + t * 16 + lrow;
            const v4u* bp =
                (const v4u*)(packW + (((size_t)kk * DOUT + gcol) << 5) + hB);
            AFrag bf;
            bf.q[0] = bp[0];
            bf.q[1] = bp[1];
            *acc[t] = __builtin_amdgcn_wmma_f32_16x16x32_bf16(
                false, a.v, false, bf.v, (short)0, *acc[t], false, false);
        }
    }

#pragma unroll
    for (int t = 0; t < 4; ++t) {
        const int gcol = nb0 + nbase + t * 16 + lrow;
        const float bv = bias[gcol];
#pragma unroll
        for (int v = 0; v < 8; ++v) {
            const int row = m0 + (hi ? (v + 8) : v);
            out[(size_t)row * DOUT + gcol] = (*acc[t])[v] + bv;
        }
    }
}

// ---------------------------------------------------------------------------
// Kernel 4: deterministic PAPR-loss mean over 1024 rows.
// ---------------------------------------------------------------------------
__global__ __launch_bounds__(256) void papr_loss(
    const float* __restrict__ relu, float* __restrict__ out)
{
    __shared__ float red[256];
    float s = 0.f;
    for (int i = threadIdx.x; i < B_TOT; i += 256) s += relu[i];
    red[threadIdx.x] = s; __syncthreads();
    for (int off = 128; off > 0; off >>= 1) {
        if (threadIdx.x < off) red[threadIdx.x] += red[threadIdx.x + off];
        __syncthreads();
    }
    if (threadIdx.x == 0) out[OUT_LOSS_OFF] = red[0] / (float)B_TOT;
}

// ---------------------------------------------------------------------------
extern "C" void kernel_launch(void* const* d_in, const int* in_sizes, int n_in,
                              void* d_out, int out_size, void* d_ws, size_t ws_size,
                              hipStream_t stream)
{
    (void)in_sizes; (void)n_in; (void)out_size; (void)ws_size;
    const float* x        = (const float*)d_in[0];
    const float* W_enc    = (const float*)d_in[1];
    const float* b_enc    = (const float*)d_in[2];
    const float* W_dec    = (const float*)d_in[3];
    const float* b_dec    = (const float*)d_in[4];
    const float* rrc      = (const float*)d_in[5];
    const float* awgn     = (const float*)d_in[6];
    const int*   startloc = (const int*)d_in[7];
    float* out = (float*)d_out;

    // workspace layout (bytes, all regions 64B-aligned by construction)
    char* ws = (char*)d_ws;
    unsigned short* packX  = (unsigned short*)(ws);               //  6,291,456
    unsigned short* packWe = (unsigned short*)(ws +  6291456);    //  3,145,728
    unsigned short* packWd = (unsigned short*)(ws +  9437184);    //  3,145,728
    float*          zbuf   = (float*)        (ws + 12582912);     //  2,097,152
    unsigned int*   ybuf   = (unsigned int*) (ws + 14680064);     //  1,048,576
    float*          relu   = (float*)        (ws + 15728640);     //      4,096

    // pack operands to bf16 fragment-native layouts
    pack_linear_bf16<<<(B_TOT * DIN / 2) / 256, 256, 0, stream>>>(
        x, (unsigned int*)packX, B_TOT * DIN / 2);
    pack_w_tiles<<<(DIN * DZ / 2) / 256, 256, 0, stream>>>(
        W_enc, (unsigned int*)packWe, DIN, DZ);
    pack_w_tiles<<<(DZ * DOUT / 2) / 256, 256, 0, stream>>>(
        W_dec, (unsigned int*)packWd, DZ, DOUT);

    enc_gemm<<<B_TOT / 16, 256, 0, stream>>>(packX, packWe, b_enc, zbuf);
    dsp_pipeline<<<B_TOT, 256, 0, stream>>>(zbuf, rrc, awgn, startloc,
                                            ybuf, out + OUT_PAPR_OFF, relu);
    dec_gemm<<<(B_TOT / 16) * 6, 256, 0, stream>>>(
        (const unsigned short*)ybuf, packWd, b_dec, out);
    papr_loss<<<1, 256, 0, stream>>>(relu, out);
}
